// WaveNet_68633577390583
// MI455X (gfx1250) — matrix operand order repaired and checked by reading, not compiled, not run on hardware
//
#include <hip/hip_runtime.h>

#define WN_T 4096
#define WN_NB 16
#define WN_L 30

typedef _Float16 v16h __attribute__((ext_vector_type(16)));
typedef _Float16 v8h  __attribute__((ext_vector_type(8)));
typedef float    v8f  __attribute__((ext_vector_type(8)));
typedef float    v4f  __attribute__((ext_vector_type(4)));

union HU { v16h v; v8h h[2]; };

// D = A(16xK f16) * B(Kx16 f16) + C, K=32, C/D f32
__device__ __forceinline__ v8f wmma32(v16h a, v16h b, v8f c) {
  return __builtin_amdgcn_wmma_f32_16x16x32_f16(false, a, false, b, (short)0, c,
                                                false, false);
}

// A fragment: lane m=lane&15 holds row m,
// halves [0:16] = K[kbase+16*hi .. +15] (contiguous, matches B fills)
__device__ __forceinline__ v16h load_afrag(const _Float16* __restrict__ W,
                                           int row_stride, int kbase, int lane) {
  const int m = lane & 15, hi = lane >> 4;
  const _Float16* p = W + (size_t)m * row_stride + kbase + 16 * hi;
  HU u;
  u.h[0] = *(const v8h*)(p);
  u.h[1] = *(const v8h*)(p + 8);
  return u.v;
}

__device__ __forceinline__ float wn_sigmoid(float x) {
  return 1.0f / (1.0f + __expf(-x));
}

// ---------------- weight prep ----------------
__global__ void wn_cvt(const float* __restrict__ in, _Float16* __restrict__ out, int n) {
  int i = blockIdx.x * 256 + threadIdx.x;
  if (i < n) out[i] = (_Float16)in[i];
}
// in: [L][O][C][2] f32 -> out0 = tap0 [L][O][C] f16, out1 = tap1
__global__ void wn_splitk(const float* __restrict__ in, _Float16* __restrict__ out0,
                          _Float16* __restrict__ out1, int n) {
  int i = blockIdx.x * 256 + threadIdx.x;
  if (i < n) {
    out0[i] = (_Float16)in[2 * i];
    out1[i] = (_Float16)in[2 * i + 1];
  }
}
__global__ void wn_zero(float* __restrict__ p, int n) {
  int i = blockIdx.x * 256 + threadIdx.x;
  if (i < n) p[i] = 0.0f;
}

// ---------------- causal 1x1 conv: cur[b][t][32] = W(32x256) @ x[b][:,t] ----------------
__global__ __launch_bounds__(64) void wn_causal(const float* __restrict__ x,
                                                const _Float16* __restrict__ w,
                                                const float* __restrict__ bias,
                                                _Float16* __restrict__ cur,
                                                int t_base, int tiles_per_b) {
  __shared__ _Float16 xt[16 * 272];  // [t][c], pad row to 272 halves
  const int tid = threadIdx.x;
  const int b = blockIdx.x / tiles_per_b;
  const int tile = blockIdx.x - b * tiles_per_b;
  const int t0 = t_base + tile * 16;

  for (int cc = tid; cc < 256; cc += 64) {
    const float* xr = x + ((size_t)b * 256 + cc) * WN_T + t0;
    __builtin_prefetch(xr + 16, 0, 1);
#pragma unroll
    for (int j = 0; j < 16; ++j) xt[j * 272 + cc] = (_Float16)xr[j];
  }
  __syncthreads();

  const int wave = tid >> 5, lane = tid & 31;
  const int r = lane & 15, hi = lane >> 4;

  v8f acc;
#pragma unroll
  for (int j = 0; j < 8; ++j) acc[j] = bias[wave * 16 + hi * 8 + j];

#pragma unroll
  for (int kb = 0; kb < 8; ++kb) {
    v16h A = load_afrag(w + wave * 16 * 256, 256, kb * 32, lane);
    HU B;
    const _Float16* bp = xt + r * 272 + kb * 32 + hi * 16;
    B.h[0] = *(const v8h*)(bp);
    B.h[1] = *(const v8h*)(bp + 8);
    acc = wmma32(A, B.v, acc);
  }

  v8h o;
#pragma unroll
  for (int j = 0; j < 8; ++j) o[j] = (_Float16)acc[j];
  *(v8h*)(cur + ((size_t)b * WN_T + t0 + r) * 32 + wave * 16 + hi * 8) = o;
}

// ---------------- one gated residual layer (one wave per 16-t tile) ----------------
__global__ __launch_bounds__(256) void wn_layer(
    const _Float16* __restrict__ cur_in, _Float16* __restrict__ cur_out,
    const _Float16* __restrict__ f0, const _Float16* __restrict__ f1,
    const _Float16* __restrict__ g0, const _Float16* __restrict__ g1,
    const _Float16* __restrict__ rw, const _Float16* __restrict__ sw,
    const float* __restrict__ fb, const float* __restrict__ gb,
    const float* __restrict__ rb, const float* __restrict__ sb,
    float* __restrict__ skip_buf, int d, int n_tiles) {
  __shared__ _Float16 zst[8 * 16 * 40];  // per-wave [t][c] staging, padded rows
  const int wave = threadIdx.x >> 5, lane = threadIdx.x & 31;
  const int gw = blockIdx.x * 8 + wave;
  if (gw >= WN_NB * n_tiles) return;  // wave-uniform exit (EXEC stays full)
  const int b = gw / n_tiles, tile = gw - b * n_tiles;
  const int t0 = WN_T - 16 * (tile + 1);
  const int r = lane & 15, hi = lane >> 4;

  const _Float16* cbase = cur_in + ((size_t)b * WN_T + t0) * 32;
  v16h Bc = *(const v16h*)(cbase + r * 32 + hi * 16);
  v16h Bp = *(const v16h*)(cbase - (size_t)d * 32 + r * 32 + hi * 16);

  v8f fa0, fa1, ga0, ga1;
#pragma unroll
  for (int j = 0; j < 8; ++j) {
    fa0[j] = fb[hi * 8 + j];      fa1[j] = fb[16 + hi * 8 + j];
    ga0[j] = gb[hi * 8 + j];      ga1[j] = gb[16 + hi * 8 + j];
  }

  // filter: W0 @ cur[t-d] + W1 @ cur[t]
  fa0 = wmma32(load_afrag(f0, 32, 0, lane), Bp, fa0);
  fa0 = wmma32(load_afrag(f1, 32, 0, lane), Bc, fa0);
  fa1 = wmma32(load_afrag(f0 + 16 * 32, 32, 0, lane), Bp, fa1);
  fa1 = wmma32(load_afrag(f1 + 16 * 32, 32, 0, lane), Bc, fa1);
  // gate
  ga0 = wmma32(load_afrag(g0, 32, 0, lane), Bp, ga0);
  ga0 = wmma32(load_afrag(g1, 32, 0, lane), Bc, ga0);
  ga1 = wmma32(load_afrag(g0 + 16 * 32, 32, 0, lane), Bp, ga1);
  ga1 = wmma32(load_afrag(g1 + 16 * 32, 32, 0, lane), Bc, ga1);

  v8f z0, z1;
#pragma unroll
  for (int j = 0; j < 8; ++j) {
    z0[j] = tanhf(fa0[j]) * wn_sigmoid(ga0[j]);
    z1[j] = tanhf(fa1[j]) * wn_sigmoid(ga1[j]);
  }

  // restage z (D layout) -> LDS [t][c] so it can be re-read as a B fragment
  _Float16* zr = zst + wave * (16 * 40);
  v8h z0h, z1h;
#pragma unroll
  for (int j = 0; j < 8; ++j) { z0h[j] = (_Float16)z0[j]; z1h[j] = (_Float16)z1[j]; }
  *(v8h*)(zr + r * 40 + hi * 8) = z0h;
  *(v8h*)(zr + r * 40 + 16 + hi * 8) = z1h;
  // same-wave LDS ops are in-order; compiler inserts s_wait_dscnt for load data
  HU bz;
  const _Float16* zp = zr + r * 40 + hi * 16;
  bz.h[0] = *(const v8h*)(zp);
  bz.h[1] = *(const v8h*)(zp + 8);

  // residual: cur_out[t] = res_w @ z + res_b + cur_in[t]
  v8h c0h = *(const v8h*)(cbase + r * 32 + hi * 8);
  v8h c1h = *(const v8h*)(cbase + r * 32 + 16 + hi * 8);
  v8f ra0, ra1;
#pragma unroll
  for (int j = 0; j < 8; ++j) {
    ra0[j] = rb[hi * 8 + j] + (float)c0h[j];
    ra1[j] = rb[16 + hi * 8 + j] + (float)c1h[j];
  }
  ra0 = wmma32(load_afrag(rw, 32, 0, lane), bz.v, ra0);
  ra1 = wmma32(load_afrag(rw + 16 * 32, 32, 0, lane), bz.v, ra1);
  v8h o0, o1;
#pragma unroll
  for (int j = 0; j < 8; ++j) { o0[j] = (_Float16)ra0[j]; o1[j] = (_Float16)ra1[j]; }
  _Float16* obase = cur_out + ((size_t)b * WN_T + t0) * 32;
  *(v8h*)(obase + r * 32 + hi * 8) = o0;
  *(v8h*)(obase + r * 32 + 16 + hi * 8) = o1;

  // skip: only the final 32 timesteps ever reach the output head
  if (t0 >= WN_T - 32) {
    float* sp = skip_buf + ((size_t)(b * 32 + (t0 - (WN_T - 32)) + r)) * 256 + hi * 8;
#pragma unroll
    for (int mt = 0; mt < 16; ++mt) {
      float* sq = sp + mt * 16;
      v4f lo = *(const v4f*)(sq), hi4 = *(const v4f*)(sq + 4);
      v8f sa;
#pragma unroll
      for (int j = 0; j < 4; ++j) {
        sa[j] = lo[j] + sb[mt * 16 + hi * 8 + j];
        sa[4 + j] = hi4[j] + sb[mt * 16 + hi * 8 + 4 + j];
      }
      sa = wmma32(load_afrag(sw + mt * 16 * 32, 32, 0, lane), bz.v, sa);
      v4f olo, ohi;
#pragma unroll
      for (int j = 0; j < 4; ++j) { olo[j] = sa[j]; ohi[j] = sa[4 + j]; }
      *(v4f*)(sq) = olo;
      *(v4f*)(sq + 4) = ohi;
    }
  }
}

// ---------------- fused output head: relu -> 256x256 -> relu -> 256x256 ----------------
__global__ __launch_bounds__(32) void wn_end(const float* __restrict__ skip_buf,
                                             const _Float16* __restrict__ w1,
                                             const float* __restrict__ b1,
                                             const _Float16* __restrict__ w2,
                                             const float* __restrict__ b2,
                                             float* __restrict__ out) {
  __shared__ _Float16 e1[16 * 264];  // [t][c] staging, padded rows
  const int lane = threadIdx.x & 31;
  const int b = blockIdx.x >> 1, tt = blockIdx.x & 1;
  const int t0 = tt * 16;
  const int r = lane & 15, hi = lane >> 4;

  v8f acc[16];
#pragma unroll
  for (int mt = 0; mt < 16; ++mt)
#pragma unroll
    for (int j = 0; j < 8; ++j) acc[mt][j] = b1[mt * 16 + hi * 8 + j];

  for (int kb = 0; kb < 8; ++kb) {
    const float* sp = skip_buf + ((size_t)(b * 32 + t0 + r)) * 256 + kb * 32 + hi * 16;
    v4f s0 = *(const v4f*)(sp), s1 = *(const v4f*)(sp + 4);
    v4f s2 = *(const v4f*)(sp + 8), s3 = *(const v4f*)(sp + 12);
    HU B;
#pragma unroll
    for (int j = 0; j < 4; ++j) {
      B.v[j]      = (_Float16)fmaxf(s0[j], 0.0f);
      B.v[4 + j]  = (_Float16)fmaxf(s1[j], 0.0f);
      B.v[8 + j]  = (_Float16)fmaxf(s2[j], 0.0f);
      B.v[12 + j] = (_Float16)fmaxf(s3[j], 0.0f);
    }
#pragma unroll
    for (int mt = 0; mt < 16; ++mt)
      acc[mt] = wmma32(load_afrag(w1 + mt * 16 * 256, 256, kb * 32, lane), B.v, acc[mt]);
  }

#pragma unroll
  for (int mt = 0; mt < 16; ++mt) {
    v8h h;
#pragma unroll
    for (int j = 0; j < 8; ++j) h[j] = (_Float16)fmaxf(acc[mt][j], 0.0f);
    *(v8h*)(e1 + r * 264 + mt * 16 + hi * 8) = h;
  }
  __syncthreads();

#pragma unroll
  for (int mt = 0; mt < 16; ++mt)
#pragma unroll
    for (int j = 0; j < 8; ++j) acc[mt][j] = b2[mt * 16 + hi * 8 + j];

  for (int kb = 0; kb < 8; ++kb) {
    HU B;
    const _Float16* zp = e1 + r * 264 + kb * 32 + hi * 16;
    B.h[0] = *(const v8h*)(zp);
    B.h[1] = *(const v8h*)(zp + 8);
#pragma unroll
    for (int mt = 0; mt < 16; ++mt)
      acc[mt] = wmma32(load_afrag(w2 + mt * 16 * 256, 256, kb * 32, lane), B.v, acc[mt]);
  }

#pragma unroll
  for (int mt = 0; mt < 16; ++mt) {
    float* op = out + ((size_t)(b * 32 + t0 + r)) * 256 + mt * 16 + hi * 8;
    v4f lo, hi4;
#pragma unroll
    for (int j = 0; j < 4; ++j) { lo[j] = acc[mt][j]; hi4[j] = acc[mt][4 + j]; }
    *(v4f*)(op) = lo;
    *(v4f*)(op + 4) = hi4;
  }
}

// ---------------- host ----------------
extern "C" void kernel_launch(void* const* d_in, const int* in_sizes, int n_in,
                              void* d_out, int out_size, void* d_ws, size_t ws_size,
                              hipStream_t stream) {
  (void)in_sizes; (void)n_in; (void)out_size; (void)ws_size;
  const float* x        = (const float*)d_in[0];
  const float* causal_w = (const float*)d_in[1];
  const float* causal_b = (const float*)d_in[2];
  const float* filt_w   = (const float*)d_in[3];
  const float* filt_b   = (const float*)d_in[4];
  const float* gate_w   = (const float*)d_in[5];
  const float* gate_b   = (const float*)d_in[6];
  const float* res_w    = (const float*)d_in[7];
  const float* res_b    = (const float*)d_in[8];
  const float* skip_w   = (const float*)d_in[9];
  const float* skip_b   = (const float*)d_in[10];
  const float* end1_w   = (const float*)d_in[11];
  const float* end1_b   = (const float*)d_in[12];
  const float* end2_w   = (const float*)d_in[13];
  const float* end2_b   = (const float*)d_in[14];
  float* out = (float*)d_out;

  // workspace layout (halves within w16 block; bytes for the rest)
  char* ws = (char*)d_ws;
  _Float16* w16 = (_Float16*)ws;
  const size_t OFF_CAUSAL = 0;        // 32*256
  const size_t OFF_FILT0  = 8192;     // 30*32*32
  const size_t OFF_FILT1  = 38912;
  const size_t OFF_GATE0  = 69632;
  const size_t OFF_GATE1  = 100352;
  const size_t OFF_RES    = 131072;   // 30*32*32
  const size_t OFF_SKIP   = 161792;   // 30*256*32
  const size_t OFF_E1     = 407552;   // 256*256
  const size_t OFF_E2     = 473088;   // 256*256
  const size_t W16_BYTES  = 538624 * 2;           // 1,077,248
  float*    skip_buf = (float*)(ws + W16_BYTES);  // 16*32*256 f32 = 524,288 B
  _Float16* cur0 = (_Float16*)(ws + W16_BYTES + 524288);
  _Float16* cur1 = cur0 + (size_t)WN_NB * WN_T * 32;  // 4 MB each

  // ---- weight prep (idempotent) ----
  wn_cvt<<<(8192 + 255) / 256, 256, 0, stream>>>(causal_w, w16 + OFF_CAUSAL, 8192);
  wn_splitk<<<(30720 + 255) / 256, 256, 0, stream>>>(filt_w, w16 + OFF_FILT0, w16 + OFF_FILT1, 30720);
  wn_splitk<<<(30720 + 255) / 256, 256, 0, stream>>>(gate_w, w16 + OFF_GATE0, w16 + OFF_GATE1, 30720);
  wn_cvt<<<(30720 + 255) / 256, 256, 0, stream>>>(res_w, w16 + OFF_RES, 30720);
  wn_cvt<<<(245760 + 255) / 256, 256, 0, stream>>>(skip_w, w16 + OFF_SKIP, 245760);
  wn_cvt<<<(65536 + 255) / 256, 256, 0, stream>>>(end1_w, w16 + OFF_E1, 65536);
  wn_cvt<<<(65536 + 255) / 256, 256, 0, stream>>>(end2_w, w16 + OFF_E2, 65536);
  wn_zero<<<(131072 + 255) / 256, 256, 0, stream>>>(skip_buf, 131072);

  // ---- trailing-window schedule ----
  int dil[WN_L];
  for (int i = 0; i < WN_L; ++i) dil[i] = 1 << (i % 10);
  int a[WN_L + 1];
  a[WN_L] = WN_T - 32;                       // 4064
  for (int i = WN_L - 1; i >= 0; --i) a[i] = a[i + 1] - dil[i];  // a[0] = 995

  // causal conv over the window (+1 slack tile for layer-0 over-compute reads)
  const int tiles0 = (WN_T - a[0] + 16 + 15) / 16;   // 195
  const int t_base = WN_T - 16 * tiles0;             // 976
  wn_causal<<<WN_NB * tiles0, 64, 0, stream>>>(x, w16 + OFF_CAUSAL, causal_b,
                                               cur0, t_base, tiles0);

  // ---- 30 dilated layers, ping-pong cur buffers ----
  _Float16* cin = cur0;
  _Float16* cout = cur1;
  for (int i = 0; i < WN_L; ++i) {
    const int n_tiles = (WN_T - a[i + 1] + 15) / 16;
    const int blocks = (WN_NB * n_tiles + 7) / 8;
    wn_layer<<<blocks, 256, 0, stream>>>(
        cin, cout,
        w16 + OFF_FILT0 + (size_t)i * 1024, w16 + OFF_FILT1 + (size_t)i * 1024,
        w16 + OFF_GATE0 + (size_t)i * 1024, w16 + OFF_GATE1 + (size_t)i * 1024,
        w16 + OFF_RES + (size_t)i * 1024, w16 + OFF_SKIP + (size_t)i * 8192,
        filt_b + (size_t)i * 32, gate_b + (size_t)i * 32,
        res_b + (size_t)i * 32, skip_b + (size_t)i * 256,
        skip_buf, dil[i], n_tiles);
    _Float16* t = cin; cin = cout; cout = t;
  }

  // ---- output head ----
  wn_end<<<32, 32, 0, stream>>>(skip_buf, w16 + OFF_E1, end1_b,
                                w16 + OFF_E2, end2_b, out);
}
